// SNN2_66898410602867
// MI455X (gfx1250) — matrix operand order repaired and checked by reading, not compiled, not run on hardware
//
#include <hip/hip_runtime.h>

typedef _Float16 v16h __attribute__((ext_vector_type(16)));
typedef _Float16 v8h  __attribute__((ext_vector_type(8)));
typedef float    v8f  __attribute__((ext_vector_type(8)));
typedef float    v4f  __attribute__((ext_vector_type(4)));

namespace {
constexpr int kT   = 100;
constexpr int kB   = 512;
constexpr int kD0  = 700;
constexpr int kD0P = 704;   // K padded to multiple of 32
constexpr int kD1  = 512;
constexpr int kD2  = 20;
constexpr int kD2P = 32;    // N padded to multiple of 16
constexpr int kM   = kT * kB;          // 51200 rows
constexpr float kAlpha = 0.8f;
constexpr float kBeta  = 0.9f;
constexpr float kThr   = 1.0f;
}

__device__ __forceinline__ v8f wmma_f16(v16h a, v16h b, v8f c) {
  // v_wmma_f32_16x16x32_f16  D = A*B + C
  return __builtin_amdgcn_wmma_f32_16x16x32_f16(false, a, false, b,
                                                (short)0, c, false, false);
}

// ---------------- weight packing (f32 -> f16, padded) ----------------

__global__ void pack_w1(const float* __restrict__ W1, _Float16* __restrict__ W1h) {
  int i = blockIdx.x * blockDim.x + threadIdx.x;          // over 512*704
  if (i >= kD1 * kD0P) return;
  int n = i / kD0P, k = i - n * kD0P;
  W1h[i] = (k < kD0) ? (_Float16)W1[n * kD0 + k] : (_Float16)0.0f;
}

__global__ void pack_w2(const float* __restrict__ W2, _Float16* __restrict__ W2h) {
  int i = blockIdx.x * blockDim.x + threadIdx.x;          // over 32*512
  if (i >= kD2P * kD1) return;
  int n = i >> 9, k = i & 511;
  W2h[i] = (n < kD2) ? (_Float16)W2[n * kD1 + k] : (_Float16)0.0f;
}

// ---------------- GEMM1: pre1[51200,512] = x[51200,700] @ W1^T ----------------
// per wave: 32M x 64N, K-loop in steps of 32 (21 full tiles + guarded tail)

__global__ __launch_bounds__(128) void gemm1(const float* __restrict__ x,
                                             const _Float16* __restrict__ W1h,
                                             float* __restrict__ pre1) {
  const int lane = threadIdx.x & 31;
  const int wave = threadIdx.x >> 5;
  const int m16  = lane & 15;
  const int hi   = lane >> 4;                 // 0 or 1
  const int nBase = blockIdx.x * 64;          // 4 n-tiles of 16
  const int mBase = (blockIdx.y * 4 + wave) * 32;  // 2 m-tiles of 16
  const int kOff  = hi * 16;                  // A-frag per-lane K shift

  v8f c[2][4];
#pragma unroll
  for (int mi = 0; mi < 2; ++mi)
#pragma unroll
    for (int ni = 0; ni < 4; ++ni)
      c[mi][ni] = (v8f){0.f, 0.f, 0.f, 0.f, 0.f, 0.f, 0.f, 0.f};

#pragma unroll 1
  for (int k0 = 0; k0 < 672; k0 += 32) {
    const int kA = k0 + kOff;
    v16h a[2];
#pragma unroll
    for (int mi = 0; mi < 2; ++mi) {
      const float* xp = x + (size_t)(mBase + mi * 16 + m16) * kD0 + kA;
      v4f p0 = *(const v4f*)(xp);           // 16B aligned: row stride 2800B
      v4f p1 = *(const v4f*)(xp + 4);
      v4f p2 = *(const v4f*)(xp + 8);
      v4f p3 = *(const v4f*)(xp + 12);
      v16h t;
#pragma unroll
      for (int j = 0; j < 4; ++j) {
        t[j]      = (_Float16)p0[j];
        t[j + 4]  = (_Float16)p1[j];
        t[j + 8]  = (_Float16)p2[j];
        t[j + 12] = (_Float16)p3[j];
      }
      a[mi] = t;
    }
#pragma unroll
    for (int ni = 0; ni < 4; ++ni) {
      // B frag: lane holds K=hi*16..hi*16+15 of column nBase+ni*16+m16 (32B aligned)
      v16h b = *(const v16h*)(W1h + (size_t)(nBase + ni * 16 + m16) * kD0P + k0 + hi * 16);
      c[0][ni] = wmma_f16(a[0], b, c[0][ni]);
      c[1][ni] = wmma_f16(a[1], b, c[1][ni]);
    }
  }

  { // K tail: k0 = 672, valid K < 700 (W1h is zero-padded, only x needs guards)
    const int k0 = 672;
    const int kA = k0 + kOff;
    v16h a[2];
#pragma unroll
    for (int mi = 0; mi < 2; ++mi) {
      const float* xrow = x + (size_t)(mBase + mi * 16 + m16) * kD0;
      v16h t;
#pragma unroll
      for (int j = 0; j < 8; ++j) {
        int k1 = kA + j;
        int k2 = kA + 8 + j;
        t[j]     = (k1 < kD0) ? (_Float16)xrow[k1] : (_Float16)0.0f;
        t[j + 8] = (k2 < kD0) ? (_Float16)xrow[k2] : (_Float16)0.0f;
      }
      a[mi] = t;
    }
#pragma unroll
    for (int ni = 0; ni < 4; ++ni) {
      v16h b = *(const v16h*)(W1h + (size_t)(nBase + ni * 16 + m16) * kD0P + k0 + hi * 16);
      c[0][ni] = wmma_f16(a[0], b, c[0][ni]);
      c[1][ni] = wmma_f16(a[1], b, c[1][ni]);
    }
  }

#pragma unroll
  for (int mi = 0; mi < 2; ++mi) {
    const int rowBase = mBase + mi * 16 + hi * 8;
#pragma unroll
    for (int ni = 0; ni < 4; ++ni) {
      float* op = pre1 + (size_t)rowBase * kD1 + nBase + ni * 16 + m16;
#pragma unroll
      for (int r = 0; r < 8; ++r) op[(size_t)r * kD1] = c[mi][ni][r];
    }
  }
}

// ---------------- scan layer 1 (sequential over T) ----------------

__global__ void scan1(const float* __restrict__ pre1, float* __restrict__ s1out,
                      _Float16* __restrict__ s1h) {
  const int tid = blockIdx.x * blockDim.x + threadIdx.x;   // < 262144 = B*512
  float syn = 0.f, mem = 0.f;
#pragma unroll 1
  for (int t = 0; t < kT; ++t) {
    const size_t idx = (size_t)t * (kB * kD1) + tid;
    float p = pre1[idx];
    syn = kAlpha * syn + p;
    float reset = (mem > kThr) ? 1.f : 0.f;               // from previous mem
    mem = (kBeta * mem + syn) * (1.f - reset);
    float spk = (mem > kThr) ? 1.f : 0.f;
    s1out[idx] = spk;
    s1h[idx]   = (_Float16)spk;
  }
}

// ---------------- GEMM2: pre2[51200,32] = s1h[51200,512] @ W2h^T ----------------
// per wave: 32M x 32N (all of padded N), K = 16 tiles of 32

__global__ __launch_bounds__(128) void gemm2(const _Float16* __restrict__ s1h,
                                             const _Float16* __restrict__ W2h,
                                             float* __restrict__ pre2) {
  const int lane = threadIdx.x & 31;
  const int wave = threadIdx.x >> 5;
  const int m16  = lane & 15;
  const int hi   = lane >> 4;
  const int mBase = (blockIdx.x * 4 + wave) * 32;
  const int kOff  = hi * 16;

  v8f c[2][2];
#pragma unroll
  for (int mi = 0; mi < 2; ++mi)
#pragma unroll
    for (int ni = 0; ni < 2; ++ni)
      c[mi][ni] = (v8f){0.f, 0.f, 0.f, 0.f, 0.f, 0.f, 0.f, 0.f};

#pragma unroll 1
  for (int k0 = 0; k0 < kD1; k0 += 32) {
    const int kA = k0 + kOff;
    v16h a[2];
#pragma unroll
    for (int mi = 0; mi < 2; ++mi) {
      const _Float16* ap = s1h + (size_t)(mBase + mi * 16 + m16) * kD1 + kA;
      v8h a0 = *(const v8h*)(ap);        // K kA..kA+7
      v8h a1 = *(const v8h*)(ap + 8);    // K kA+8..kA+15
      v16h t;
#pragma unroll
      for (int j = 0; j < 8; ++j) { t[j] = a0[j]; t[j + 8] = a1[j]; }
      a[mi] = t;
    }
#pragma unroll
    for (int ni = 0; ni < 2; ++ni) {
      v16h b = *(const v16h*)(W2h + (size_t)(ni * 16 + m16) * kD1 + k0 + hi * 16);
      c[0][ni] = wmma_f16(a[0], b, c[0][ni]);
      c[1][ni] = wmma_f16(a[1], b, c[1][ni]);
    }
  }

#pragma unroll
  for (int mi = 0; mi < 2; ++mi) {
    const int rowBase = mBase + mi * 16 + hi * 8;
#pragma unroll
    for (int ni = 0; ni < 2; ++ni) {
      float* op = pre2 + (size_t)rowBase * kD2P + ni * 16 + m16;
#pragma unroll
      for (int r = 0; r < 8; ++r) op[(size_t)r * kD2P] = c[mi][ni][r];
    }
  }
}

// ---------------- scan layer 2: writes s2 twice (tuple (s2, s1, s2)) ----------------

__global__ void scan2(const float* __restrict__ pre2, float* __restrict__ outA,
                      float* __restrict__ outB) {
  const int tid = blockIdx.x * blockDim.x + threadIdx.x;   // < 10240 = B*20
  if (tid >= kB * kD2) return;
  const int b = tid / kD2;
  const int f = tid - b * kD2;
  float syn = 0.f, mem = 0.f;
#pragma unroll 1
  for (int t = 0; t < kT; ++t) {
    float p = pre2[((size_t)t * kB + b) * kD2P + f];
    syn = kAlpha * syn + p;
    float reset = (mem > kThr) ? 1.f : 0.f;
    mem = (kBeta * mem + syn) * (1.f - reset);
    float spk = (mem > kThr) ? 1.f : 0.f;
    const size_t o = (size_t)t * (kB * kD2) + tid;
    outA[o] = spk;
    outB[o] = spk;
  }
}

// ---------------- host launcher ----------------

extern "C" void kernel_launch(void* const* d_in, const int* in_sizes, int n_in,
                              void* d_out, int out_size, void* d_ws, size_t ws_size,
                              hipStream_t stream) {
  const float* x  = (const float*)d_in[0];   // [T,B,700]
  const float* W1 = (const float*)d_in[1];   // [512,700]
  const float* W2 = (const float*)d_in[2];   // [20,512]

  // workspace layout
  char* ws = (char*)d_ws;
  _Float16* W1h = (_Float16*)(ws);                         //   720,896 B
  _Float16* W2h = (_Float16*)(ws + 720896);                //    32,768 B
  float*    pre1 = (float*)(ws + 753664);                  // 104,857,600 B
  _Float16* s1h  = (_Float16*)(ws + 105611264);            //  52,428,800 B
  float*    pre2 = (float*)(ws + 158040064);               //   6,553,600 B

  float* out  = (float*)d_out;
  float* s2a  = out;                                   // [T,B,20]
  float* s1o  = out + (size_t)kT * kB * kD2;           // [T,B,512]
  float* s2b  = out + (size_t)kT * kB * (kD2 + kD1);   // [T,B,20]

  pack_w1<<<(kD1 * kD0P + 255) / 256, 256, 0, stream>>>(W1, W1h);
  pack_w2<<<(kD2P * kD1 + 255) / 256, 256, 0, stream>>>(W2, W2h);

  // GEMM1: grid.x sweeps N (8 x 64), grid.y sweeps M (400 x 128) -> x reused via L2
  gemm1<<<dim3(8, 400), 128, 0, stream>>>(x, W1h, pre1);

  scan1<<<(kB * kD1) / 256, 256, 0, stream>>>(pre1, s1o, s1h);

  gemm2<<<kM / 128, 128, 0, stream>>>(s1h, W2h, pre2);

  scan2<<<(kB * kD2 + 255) / 256, 256, 0, stream>>>(pre2, s2a, s2b);
}
